// EpisodeMultiheadAttentionBlock_24945170055869
// MI455X (gfx1250) — matrix operand order repaired and checked by reading, not compiled, not run on hardware
//
#include <hip/hip_runtime.h>
#include <hip/hip_bf16.h>

// ---------------------------------------------------------------------------
// EpisodeMultiheadAttentionBlock for MI455X (gfx1250, wave32, WMMA bf16,
// async global->LDS staging, double-buffered GEMM pipeline)
// ---------------------------------------------------------------------------

typedef unsigned short u16;
typedef __attribute__((ext_vector_type(16))) __bf16 v16bf;
typedef __attribute__((ext_vector_type(8)))  float  v8f;
typedef __attribute__((ext_vector_type(4)))  int    v4i;

#define BB 8
#define SS 1024
#define EE 1024
#define HH 8
#define HD 128
#define MM (BB * SS)   // 8192 rows

#define AS1 __attribute__((address_space(1)))
#define AS3 __attribute__((address_space(3)))

#if defined(__has_builtin)
#  if __has_builtin(__builtin_amdgcn_global_load_async_to_lds_b128)
#    define HAVE_ASYNC_LDS 1
#  endif
#  if __has_builtin(__builtin_amdgcn_s_wait_asynccnt)
#    define HAVE_ASYNC_WAIT 1
#  endif
#endif

union FragU { v16bf v; uint4 q[2]; u16 s[16]; };
union AccU  { v8f v; float f[8]; };
union C16   { uint4 q; u16 s[8]; };

__device__ __forceinline__ u16 f2bf(float f) {
    unsigned u = __float_as_uint(f);
    unsigned r = u + 0x7FFFu + ((u >> 16) & 1u);   // round-to-nearest-even
    return (u16)(r >> 16);
}

__device__ __forceinline__ v8f wmma_bf16(const FragU& a, const FragU& b, v8f c) {
    return __builtin_amdgcn_wmma_f32_16x16x32_bf16(
        false, a.v, false, b.v, (short)0, c, false, false);
}

// 16B global -> LDS copy: async on CDNA5 (ASYNCcnt), sync fallback otherwise
__device__ __forceinline__ void cp16(u16* lds, const u16* g) {
#ifdef HAVE_ASYNC_LDS
    __builtin_amdgcn_global_load_async_to_lds_b128(
        (AS1 v4i*)g, (AS3 v4i*)lds, 0, 0);
#else
    *(uint4*)lds = *(const uint4*)g;
#endif
}
__device__ __forceinline__ void cp16_wait() {
#if defined(HAVE_ASYNC_LDS) && defined(HAVE_ASYNC_WAIT)
    __builtin_amdgcn_s_wait_asynccnt(0);
#endif
}

__device__ __forceinline__ float halfmax(float v) {
    v = fmaxf(v, __shfl_xor(v, 1, 32));
    v = fmaxf(v, __shfl_xor(v, 2, 32));
    v = fmaxf(v, __shfl_xor(v, 4, 32));
    v = fmaxf(v, __shfl_xor(v, 8, 32));
    return v;
}
__device__ __forceinline__ float halfsum(float v) {
    v += __shfl_xor(v, 1, 32);
    v += __shfl_xor(v, 2, 32);
    v += __shfl_xor(v, 4, 32);
    v += __shfl_xor(v, 8, 32);
    return v;
}

// ---------------------------------------------------------------------------
// fp32 -> bf16 cast (4 elems / thread)
// ---------------------------------------------------------------------------
__global__ __launch_bounds__(256)
void cast_bf16_kernel(const float* __restrict__ in, u16* __restrict__ out, int n) {
    int i = (blockIdx.x * 256 + threadIdx.x) * 4;
    if (i + 3 < n) {
        float4 v = *(const float4*)(in + i);
        out[i + 0] = f2bf(v.x);
        out[i + 1] = f2bf(v.y);
        out[i + 2] = f2bf(v.z);
        out[i + 3] = f2bf(v.w);
    }
}

// ---------------------------------------------------------------------------
// Generic bf16 GEMM:  C[M,N] = act( A1*B1 (+ A2*B2) + bias )
//   Tile 128x128, BK=32, 256 threads = 8 waves; wave -> 32x64 subtile.
//   Double-buffered LDS; A staged via async global->LDS; B staged transposed.
//   ACT: 0 none, 1 relu, 2 sigmoid, 3 tanh
// ---------------------------------------------------------------------------
#define GLDK 40   // LDS k-stride (32 + 8 pad), multiple of 8 for 16B alignment

template <int ACT, bool DUAL, bool OUTF, bool OUTB>
__global__ __launch_bounds__(256)
void gemm_bf16_kernel(const u16* __restrict__ A, const u16* __restrict__ Bw,
                      const u16* __restrict__ A2, const u16* __restrict__ B2,
                      const float* __restrict__ bias,
                      float* __restrict__ Cf, u16* __restrict__ Cb,
                      int M, int N, int K) {
    __shared__ __align__(16) u16 As[2][128 * GLDK];   // [m][k]
    __shared__ __align__(16) u16 BsT[2][128 * GLDK];  // [n][k]

    const int tid  = threadIdx.x;
    const int w    = tid >> 5;
    const int lane = tid & 31;
    const int half = lane >> 4;
    const int ln   = lane & 15;
    const int m0   = blockIdx.y * 128;
    const int n0   = blockIdx.x * 128;
    const int mBase = (w & 3) * 32;
    const int nBase = (w >> 2) * 64;

    const int KT = K / 32;
    const int T  = (DUAL ? 2 : 1) * KT;

    // per-thread staging chunk coordinates (2 x 16B chunks per array)
    int ar[2], ac[2], br[2], bc[2];
#pragma unroll
    for (int i = 0; i < 2; ++i) {
        int c = tid + i * 256;
        ar[i] = c >> 2;  ac[i] = (c & 3) * 8;    // A: 128 rows x 4 chunks
        br[i] = c >> 4;  bc[i] = (c & 15) * 8;   // B: 32 rows x 16 chunks
    }

    auto srcA = [&](int t) { return (DUAL && t >= KT) ? A2 : A; };
    auto srcB = [&](int t) { return (DUAL && t >= KT) ? B2 : Bw; };
    auto kof  = [&](int t) { return ((DUAL && t >= KT) ? t - KT : t) * 32; };

    auto issueA = [&](int t, int buf) {
        const u16* Ap = srcA(t);
        int k0 = kof(t);
#pragma unroll
        for (int i = 0; i < 2; ++i)
            cp16(&As[buf][ar[i] * GLDK + ac[i]],
                 Ap + (size_t)(m0 + ar[i]) * K + k0 + ac[i]);
    };
    auto loadB = [&](int t, uint4* regs) {
        const u16* Bp = srcB(t);
        int k0 = kof(t);
#pragma unroll
        for (int i = 0; i < 2; ++i)
            regs[i] = *(const uint4*)(Bp + (size_t)(k0 + br[i]) * N + n0 + bc[i]);
    };
    auto storeB = [&](const uint4* regs, int buf) {
#pragma unroll
        for (int i = 0; i < 2; ++i) {
            C16 t; t.q = regs[i];
#pragma unroll
            for (int j = 0; j < 8; ++j)
                BsT[buf][(bc[i] + j) * GLDK + br[i]] = t.s[j];
        }
    };

    v8f acc[2][4];
    const v8f vzero = {0.f, 0.f, 0.f, 0.f, 0.f, 0.f, 0.f, 0.f};
#pragma unroll
    for (int mi = 0; mi < 2; ++mi)
#pragma unroll
        for (int ni = 0; ni < 4; ++ni) acc[mi][ni] = vzero;

    // ---- prologue: stage tile 0 into buffer 0 ----
    uint4 bregs[2];
    issueA(0, 0);
    loadB(0, bregs);
    storeB(bregs, 0);
    cp16_wait();
    __syncthreads();

    // ---- pipelined main loop: 1 barrier / iteration ----
    for (int t = 0; t < T; ++t) {
        const int cur = t & 1, nxt = cur ^ 1;
        const bool more = (t + 1 < T);
        if (more) {                         // overlap next tile with compute
            issueA(t + 1, nxt);
            loadB(t + 1, bregs);
        }
        if (t + 2 < T) {                    // L2 prefetch two tiles ahead
            __builtin_prefetch(srcA(t + 2) + (size_t)(m0 + ar[0]) * K + kof(t + 2) + ac[0], 0, 0);
            __builtin_prefetch(srcB(t + 2) + (size_t)(kof(t + 2) + br[0]) * N + n0 + bc[0], 0, 0);
        }

        FragU afr[2], bfr[4];
#pragma unroll
        for (int mi = 0; mi < 2; ++mi) {
            const u16* p = &As[cur][(mBase + mi * 16 + ln) * GLDK + half * 8];
            afr[mi].q[0] = *(const uint4*)p;
            afr[mi].q[1] = *(const uint4*)(p + 16);
        }
#pragma unroll
        for (int ni = 0; ni < 4; ++ni) {
            const u16* p = &BsT[cur][(nBase + ni * 16 + ln) * GLDK + half * 8];
            bfr[ni].q[0] = *(const uint4*)p;
            bfr[ni].q[1] = *(const uint4*)(p + 16);
        }
#pragma unroll
        for (int mi = 0; mi < 2; ++mi)
#pragma unroll
            for (int ni = 0; ni < 4; ++ni)
                acc[mi][ni] = wmma_bf16(afr[mi], bfr[ni], acc[mi][ni]);

        if (more) storeB(bregs, nxt);
        cp16_wait();
        __syncthreads();
    }

    // ---- epilogue (branch-free, template-folded) ----
#pragma unroll
    for (int mi = 0; mi < 2; ++mi) {
#pragma unroll
        for (int ni = 0; ni < 4; ++ni) {
            AccU u; u.v = acc[mi][ni];
            int col = n0 + nBase + ni * 16 + ln;
            float bv = bias ? bias[col] : 0.0f;
#pragma unroll
            for (int r = 0; r < 8; ++r) {
                int row = m0 + mBase + mi * 16 + r + 8 * half;
                float v = u.f[r] + bv;
                if (ACT == 1) v = fmaxf(v, 0.0f);
                if (ACT == 2) v = 1.0f / (1.0f + __expf(-v));
                if (ACT == 3) v = tanhf(v);
                size_t idx = (size_t)row * N + col;
                if (OUTF) Cf[idx] = v;
                if (OUTB) Cb[idx] = f2bf(v);
            }
        }
    }
}

// ---------------------------------------------------------------------------
// RoPE over full embed dim (interleaved pairs), fp32 in -> bf16 out
// ---------------------------------------------------------------------------
__global__ __launch_bounds__(256)
void rope_kernel(const float* __restrict__ in, const int* __restrict__ idx,
                 u16* __restrict__ out) {
    int bs = blockIdx.x;
    float pos = (float)idx[bs];
    const float* src = in + (size_t)bs * EE;
    u16* dst = out + (size_t)bs * EE;
#pragma unroll
    for (int j = 0; j < 2; ++j) {
        int p = threadIdx.x + j * 256;                       // pair 0..511
        float inv = __expf(-((float)(2 * p) / (float)EE) * 9.210340371976184f);
        float f = pos * inv;
        float s, c;
        __sincosf(f, &s, &c);
        float x0 = src[2 * p], x1 = src[2 * p + 1];
        dst[2 * p]     = f2bf(x0 * c - x1 * s);
        dst[2 * p + 1] = f2bf(x0 * s + x1 * c);
    }
}

// ---------------------------------------------------------------------------
// Causal flash attention, bf16 in/out, fp32 online softmax.
// block = 128 threads (4 waves); block handles (b, h, 64-query tile)
// ---------------------------------------------------------------------------
#define QLD 136   // 128 + 8 pad
#define VLD 72    // 64 + 8 pad

__global__ __launch_bounds__(128)
void attn_kernel(const u16* __restrict__ Qb, const u16* __restrict__ Kb,
                 const u16* __restrict__ Vb, u16* __restrict__ Ob) {
    __shared__ __align__(16) u16 Qs[64 * QLD];
    __shared__ __align__(16) u16 Ks[64 * QLD];
    __shared__ __align__(16) u16 VsT[128 * VLD];   // [d][kv]
    __shared__ __align__(16) u16 Ps[4 * 16 * 64];  // per-wave P staging

    const int tid  = threadIdx.x;
    const int w    = tid >> 5;
    const int lane = tid & 31;
    const int half = lane >> 4;
    const int ln   = lane & 15;
    const int q0 = blockIdx.x * 64;
    const int h  = blockIdx.y;
    const int b  = blockIdx.z;
    const size_t headOff = (size_t)h * HD;

    // --- stage Q tile 64x128 via async path ---
#pragma unroll
    for (int i = 0; i < 8; ++i) {
        int c = tid + i * 128;                     // 1024 chunks
        int r = c >> 4, cc = (c & 15) * 8;
        cp16(&Qs[r * QLD + cc],
             Qb + (size_t)(b * SS + q0 + r) * EE + headOff + cc);
    }
    cp16_wait();
    __syncthreads();

    FragU qf[4];
#pragma unroll
    for (int kk = 0; kk < 4; ++kk) {
        const u16* p = &Qs[(w * 16 + ln) * QLD + kk * 32 + half * 8];
        qf[kk].q[0] = *(const uint4*)p;
        qf[kk].q[1] = *(const uint4*)(p + 16);
    }

    const v8f vzero = {0.f, 0.f, 0.f, 0.f, 0.f, 0.f, 0.f, 0.f};
    float mrow[8], lrow[8];
    v8f oacc[8];
#pragma unroll
    for (int r = 0; r < 8; ++r) { mrow[r] = -1e30f; lrow[r] = 0.0f; }
#pragma unroll
    for (int ni = 0; ni < 8; ++ni) oacc[ni] = vzero;

    const float scale = 0.08838834764831845f;      // 1/sqrt(128)
    const int nt = q0 / 64 + 1;

    for (int kt = 0; kt < nt; ++kt) {
        const int kv0 = kt * 64;
        __syncthreads();
        // --- stage K (async) and V (transposed, manual) ---
#pragma unroll
        for (int i = 0; i < 8; ++i) {
            int c = tid + i * 128;
            int r = c >> 4, cc = (c & 15) * 8;
            cp16(&Ks[r * QLD + cc],
                 Kb + (size_t)(b * SS + kv0 + r) * EE + headOff + cc);
        }
#pragma unroll
        for (int i = 0; i < 8; ++i) {
            int c = tid + i * 128;
            int r = c >> 4, cc = (c & 15) * 8;
            C16 t;
            t.q = *(const uint4*)(Vb + (size_t)(b * SS + kv0 + r) * EE + headOff + cc);
#pragma unroll
            for (int j = 0; j < 8; ++j) VsT[(cc + j) * VLD + r] = t.s[j];
        }
        cp16_wait();
        __syncthreads();

        // --- S = Q K^T : 16x64 per wave ---
        AccU sc[4];
#pragma unroll
        for (int ni = 0; ni < 4; ++ni) {
            v8f a = vzero;
#pragma unroll
            for (int kk = 0; kk < 4; ++kk) {
                FragU kf;
                const u16* p = &Ks[(ni * 16 + ln) * QLD + kk * 32 + half * 8];
                kf.q[0] = *(const uint4*)p;
                kf.q[1] = *(const uint4*)(p + 16);
                a = wmma_bf16(qf[kk], kf, a);
            }
            sc[ni].v = a;
        }

        // --- scale + causal mask (diagonal tile only) ---
        const bool diag = (kv0 == q0);
#pragma unroll
        for (int ni = 0; ni < 4; ++ni)
#pragma unroll
            for (int r = 0; r < 8; ++r) {
                float v = sc[ni].f[r] * scale;
                if (diag) {
                    int qrow = w * 16 + r + 8 * half;
                    int kcol = ni * 16 + ln;
                    if (kcol > qrow) v = -1e30f;
                }
                sc[ni].f[r] = v;
            }

        // --- online softmax ---
        float alpha[8];
#pragma unroll
        for (int r = 0; r < 8; ++r) {
            float mx = fmaxf(fmaxf(sc[0].f[r], sc[1].f[r]),
                             fmaxf(sc[2].f[r], sc[3].f[r]));
            mx = halfmax(mx);
            float mnew = fmaxf(mrow[r], mx);
            alpha[r] = __expf(mrow[r] - mnew);
            mrow[r] = mnew;
        }
        float rs[8];
#pragma unroll
        for (int r = 0; r < 8; ++r) rs[r] = 0.0f;
#pragma unroll
        for (int ni = 0; ni < 4; ++ni)
#pragma unroll
            for (int r = 0; r < 8; ++r) {
                float p = __expf(sc[ni].f[r] - mrow[r]);
                rs[r] += p;
                Ps[w * 1024 + (r + 8 * half) * 64 + ni * 16 + ln] = f2bf(p);
            }
#pragma unroll
        for (int r = 0; r < 8; ++r) lrow[r] = lrow[r] * alpha[r] + halfsum(rs[r]);

        // rescale O by alpha (per owned row)
#pragma unroll
        for (int ni = 0; ni < 8; ++ni) {
            AccU u; u.v = oacc[ni];
#pragma unroll
            for (int r = 0; r < 8; ++r) u.f[r] *= alpha[r];
            oacc[ni] = u.v;
        }

        // --- O += P V ---
        FragU pf[2];
#pragma unroll
        for (int kk = 0; kk < 2; ++kk) {
            const u16* p = &Ps[w * 1024 + ln * 64 + kk * 32 + half * 8];
            pf[kk].q[0] = *(const uint4*)p;
            pf[kk].q[1] = *(const uint4*)(p + 16);
        }
#pragma unroll
        for (int ni = 0; ni < 8; ++ni) {
#pragma unroll
            for (int kk = 0; kk < 2; ++kk) {
                FragU vf;
                const u16* p = &VsT[(ni * 16 + ln) * VLD + kk * 32 + half * 8];
                vf.q[0] = *(const uint4*)p;
                vf.q[1] = *(const uint4*)(p + 16);
                oacc[ni] = wmma_bf16(pf[kk], vf, oacc[ni]);
            }
        }
    }

    // --- normalize + write ctx (bf16) ---
#pragma unroll
    for (int ni = 0; ni < 8; ++ni) {
        AccU u; u.v = oacc[ni];
#pragma unroll
        for (int r = 0; r < 8; ++r) {
            int qrow = q0 + w * 16 + r + 8 * half;
            float v = u.f[r] / lrow[r];
            Ob[(size_t)(b * SS + qrow) * EE + headOff + ni * 16 + ln] = f2bf(v);
        }
    }
}

// ---------------------------------------------------------------------------
// rx = bf16(r * x)
// ---------------------------------------------------------------------------
__global__ __launch_bounds__(256)
void mulcast_kernel(const float* __restrict__ r, const float* __restrict__ x,
                    u16* __restrict__ o, int n) {
    int i = blockIdx.x * 256 + threadIdx.x;
    if (i < n) o[i] = f2bf(r[i] * x[i]);
}

// ---------------------------------------------------------------------------
// out = (1 - z) * x + z * h
// ---------------------------------------------------------------------------
__global__ __launch_bounds__(256)
void gate_kernel(const float* __restrict__ x, const float* __restrict__ z,
                 const float* __restrict__ h, float* __restrict__ out, int n) {
    int i = blockIdx.x * 256 + threadIdx.x;
    if (i < n) {
        float zz = z[i];
        out[i] = (1.0f - zz) * x[i] + zz * h[i];
    }
}

// ---------------------------------------------------------------------------
// Host orchestration
// ---------------------------------------------------------------------------
extern "C" void kernel_launch(void* const* d_in, const int* in_sizes, int n_in,
                              void* d_out, int out_size, void* d_ws, size_t ws_size,
                              hipStream_t stream) {
    const float* x    = (const float*)d_in[0];
    const int*   qidx = (const int*)d_in[1];
    const int*   kidx = (const int*)d_in[2];
    const float* bq   = (const float*)d_in[4];
    const float* bk   = (const float*)d_in[6];
    const float* bv   = (const float*)d_in[8];
    const float* bo   = (const float*)d_in[10];
    const float* bxz  = (const float*)d_in[14];
    // weight order: Wq, Wk, Wv, Wo, Wxr, Wyr, Wxz, Wyz, Wxg, Wyg
    const float* Wsrc[10] = {
        (const float*)d_in[3],  (const float*)d_in[5],  (const float*)d_in[7],
        (const float*)d_in[9],  (const float*)d_in[11], (const float*)d_in[12],
        (const float*)d_in[13], (const float*)d_in[15], (const float*)d_in[16],
        (const float*)d_in[17]
    };

    char* ws = (char*)d_ws;
    size_t off = 0;
    auto carve = [&](size_t bytes) -> void* {
        void* p = ws + off;
        off += (bytes + 255) & ~(size_t)255;
        return p;
    };
    const size_t nAct = (size_t)MM * EE;  // 8,388,608
    const size_t nW   = (size_t)EE * EE;  // 1,048,576

    u16* xb = (u16*)carve(nAct * 2);
    u16* wb[10];
    for (int i = 0; i < 10; ++i) wb[i] = (u16*)carve(nW * 2);
    float* qf   = (float*)carve(nAct * 4);
    float* kf   = (float*)carve(nAct * 4);
    u16*   qbf  = (u16*)carve(nAct * 2);
    u16*   kbf  = (u16*)carve(nAct * 2);
    u16*   vbf  = (u16*)carve(nAct * 2);
    u16*   ctxb = (u16*)carve(nAct * 2);
    u16*   yb   = (u16*)carve(nAct * 2);
    float* rf   = (float*)carve(nAct * 4);
    float* zf   = (float*)carve(nAct * 4);
    float* hf   = (float*)carve(nAct * 4);
    u16*   rxb  = (u16*)carve(nAct * 2);

    // ---- casts to bf16 ----
    cast_bf16_kernel<<<(int)(nAct / 1024), 256, 0, stream>>>(x, xb, (int)nAct);
    for (int i = 0; i < 10; ++i)
        cast_bf16_kernel<<<(int)(nW / 1024), 256, 0, stream>>>(Wsrc[i], wb[i], (int)nW);

    const dim3 gGrid(EE / 128, MM / 128);   // (8, 64)

    // ---- QKV projections ----
    gemm_bf16_kernel<0, false, true, false><<<gGrid, 256, 0, stream>>>(
        xb, wb[0], nullptr, nullptr, bq, qf, nullptr, MM, EE, EE);
    gemm_bf16_kernel<0, false, true, false><<<gGrid, 256, 0, stream>>>(
        xb, wb[1], nullptr, nullptr, bk, kf, nullptr, MM, EE, EE);
    gemm_bf16_kernel<0, false, false, true><<<gGrid, 256, 0, stream>>>(
        xb, wb[2], nullptr, nullptr, bv, nullptr, vbf, MM, EE, EE);

    // ---- RoPE on q,k (fp32 -> bf16) ----
    rope_kernel<<<MM, 256, 0, stream>>>(qf, qidx, qbf);
    rope_kernel<<<MM, 256, 0, stream>>>(kf, kidx, kbf);

    // ---- causal flash attention ----
    attn_kernel<<<dim3(SS / 64, HH, BB), 128, 0, stream>>>(qbf, kbf, vbf, ctxb);

    // ---- out proj + ReLU ----
    gemm_bf16_kernel<1, false, false, true><<<gGrid, 256, 0, stream>>>(
        ctxb, wb[3], nullptr, nullptr, bo, nullptr, yb, MM, EE, EE);

    // ---- r = sigmoid(x Wxr + y Wyr) ----
    gemm_bf16_kernel<2, true, true, false><<<gGrid, 256, 0, stream>>>(
        xb, wb[4], yb, wb[5], nullptr, rf, nullptr, MM, EE, EE);
    mulcast_kernel<<<(int)(nAct / 256), 256, 0, stream>>>(rf, x, rxb, (int)nAct);

    // ---- z = sigmoid(x Wxz + bxz + y Wyz) ----
    gemm_bf16_kernel<2, true, true, false><<<gGrid, 256, 0, stream>>>(
        xb, wb[6], yb, wb[7], bxz, zf, nullptr, MM, EE, EE);

    // ---- h = tanh((r*x) Wxg + y Wyg) ----
    gemm_bf16_kernel<3, true, true, false><<<gGrid, 256, 0, stream>>>(
        rxb, wb[8], yb, wb[9], nullptr, hf, nullptr, MM, EE, EE);

    // ---- out = (1-z) x + z h ----
    gate_kernel<<<(int)(nAct / 256), 256, 0, stream>>>(x, zf, hf, (float*)d_out,
                                                       (int)nAct);
}
